// Net_77275051589684
// MI455X (gfx1250) — matrix operand order repaired and checked by reading, not compiled, not run on hardware
//
#include <hip/hip_runtime.h>
#include <hip/hip_bf16.h>

// ---------------------------------------------------------------------------
// Types for CDNA5 WMMA
// ---------------------------------------------------------------------------
typedef __attribute__((ext_vector_type(16))) __bf16 bf16x16;
typedef __attribute__((ext_vector_type(4)))  __bf16 bf16x4;
typedef __attribute__((ext_vector_type(8)))  float  floatx8;

#define N_NODES_C   20000
#define E0_EDGES_C  160000
#define ET_EDGES_C  180000   // + self loops
#define N_GRAPHS_C  64
#define NEG_SLOPE_C 0.2f

// ---------------------------------------------------------------------------
// Helpers
// ---------------------------------------------------------------------------
__device__ __forceinline__ unsigned short f32_to_bf16_bits(float f) {
  unsigned u = __float_as_uint(f);
  unsigned r = 0x7FFFu + ((u >> 16) & 1u);   // round-to-nearest-even
  return (unsigned short)((u + r) >> 16);
}

// order-preserving float -> uint key (for atomicMax-based segment max)
__device__ __forceinline__ unsigned fkey(float f) {
  unsigned u = __float_as_uint(f);
  return (u & 0x80000000u) ? ~u : (u | 0x80000000u);
}
__device__ __forceinline__ float funkey(unsigned k) {
  unsigned u = (k & 0x80000000u) ? (k & 0x7FFFFFFFu) : ~k;
  return __uint_as_float(u);
}

__device__ __forceinline__ int edge_src(const int* ei, int e) {
  return (e < E0_EDGES_C) ? ei[e] : (e - E0_EDGES_C);
}
__device__ __forceinline__ int edge_dst(const int* ei, int e) {
  return (e < E0_EDGES_C) ? ei[E0_EDGES_C + e] : (e - E0_EDGES_C);
}

// ---------------------------------------------------------------------------
// bf16 WMMA GEMM:  C[M,N] = act(A[M,K]) @ W,  with Bt[n,k] = bf16(W[k,n])
// Block tile 128x128, 8 waves (4x2), each wave 32x64 via 2x4 WMMA frags.
// Register double-buffering: tile k+1 global loads issue while tile k's
// ds_load + v_wmma stream executes.
// ---------------------------------------------------------------------------
#define BM 128
#define BN 128
#define LDA 40   // half-words per LDS row (conflict-free, 16B aligned rows)
#define LDB 40

__global__ __launch_bounds__(256) void gemm_bf16_wmma(
    const float* __restrict__ A, const unsigned short* __restrict__ Bt,
    float* __restrict__ C, int M, int N, int K, int Kp, int reluA)
{
  __shared__ unsigned short sA[BM * LDA];
  __shared__ unsigned short sB[BN * LDB];

  const int tid  = threadIdx.x;
  const int wave = tid >> 5;
  const int lane = tid & 31;
  const int l    = lane & 15;
  const int hi   = lane >> 4;
  const int wm   = wave >> 1;           // 0..3  (M direction, 32 rows each)
  const int wn   = wave & 1;            // 0..1  (N direction, 64 cols each)
  const int m0   = blockIdx.x * BM;
  const int n0   = blockIdx.y * BN;

  // Per-thread staging coordinates (A: 4 float4 chunks, B: 2 uint4 chunks)
  int aRow[4], aKK[4], aRowC[4];
  for (int i = 0; i < 4; ++i) {
    int q   = tid + i * 256;            // 0..1023
    aRow[i] = q >> 3;                   // 0..127
    aKK[i]  = (q & 7) << 2;             // 0,4,...,28
    int g   = m0 + aRow[i];
    aRowC[i] = (g < M) ? g : (M - 1);   // clamped row (OOB rows zeroed on store)
  }
  int bRow[2], bKK[2];
  for (int i = 0; i < 2; ++i) {
    int q   = tid + i * 256;            // 0..511
    bRow[i] = q >> 2;                   // 0..127
    bKK[i]  = (q & 3) << 3;             // 0,8,16,24
  }

  auto loadTile = [&](int kc, float4 (&ar)[4], uint4 (&br)[2]) {
    for (int i = 0; i < 4; ++i) {
      int gk = kc + aKK[i];
      int kclamp = ((gk + 4) <= K) ? gk : 0;
      ar[i] = *(const float4*)(A + (size_t)aRowC[i] * K + kclamp);
    }
    for (int i = 0; i < 2; ++i)
      br[i] = *(const uint4*)(Bt + (size_t)(n0 + bRow[i]) * Kp + kc + bKK[i]);
  };

  floatx8 acc[2][4];
  for (int i = 0; i < 2; ++i)
    for (int j = 0; j < 4; ++j)
      for (int v = 0; v < 8; ++v) acc[i][j][v] = 0.0f;

  float4 aReg[4];
  uint4  bReg[2];
  loadTile(0, aReg, bReg);              // prologue: tile 0

  for (int kc = 0; kc < Kp; kc += 32) {
    __syncthreads();                    // prior tile's frag reads complete
    // --- commit staged registers to LDS (fp32 -> bf16, zero OOB) ---
    for (int i = 0; i < 4; ++i) {
      float4 v = aReg[i];
      bool ok = (m0 + aRow[i] < M) && ((kc + aKK[i] + 4) <= K);
      if (reluA) {
        v.x = fmaxf(v.x, 0.f); v.y = fmaxf(v.y, 0.f);
        v.z = fmaxf(v.z, 0.f); v.w = fmaxf(v.w, 0.f);
      }
      if (!ok) { v.x = 0.f; v.y = 0.f; v.z = 0.f; v.w = 0.f; }
      bf16x4 pk = { (__bf16)v.x, (__bf16)v.y, (__bf16)v.z, (__bf16)v.w };
      *(bf16x4*)&sA[aRow[i] * LDA + aKK[i]] = pk;   // 8B aligned
    }
    for (int i = 0; i < 2; ++i)
      *(uint4*)&sB[bRow[i] * LDB + bKK[i]] = bReg[i];
    __syncthreads();

    // --- prefetch next tile into registers (overlaps WMMA below) ---
    int kn = kc + 32;
    if (kn < Kp) loadTile(kn, aReg, bReg);

    // --- load fragments (ISA 16-bit A 16x32 / B 32x16 layouts) + WMMA ---
    bf16x16 afrag[2], bfrag[4];
    for (int fm = 0; fm < 2; ++fm) {
      int base = (wm * 32 + fm * 16 + l) * LDA + hi * 8;
      ((uint4*)&afrag[fm])[0] = *(const uint4*)&sA[base];        // K = hi*8 + 0..7
      ((uint4*)&afrag[fm])[1] = *(const uint4*)&sA[base + 16];   // K = 16 + hi*8 + 0..7
    }
    for (int fn = 0; fn < 4; ++fn) {
      int base = (wn * 64 + fn * 16 + l) * LDB + hi * 8;
      ((uint4*)&bfrag[fn])[0] = *(const uint4*)&sB[base];        // K = hi*8 + 0..7
      ((uint4*)&bfrag[fn])[1] = *(const uint4*)&sB[base + 16];   // K = 16 + hi*8 + 0..7
    }
    for (int fm = 0; fm < 2; ++fm)
      for (int fn = 0; fn < 4; ++fn)
        acc[fm][fn] = __builtin_amdgcn_wmma_f32_16x16x32_bf16(
            false, afrag[fm], false, bfrag[fn], (short)0, acc[fm][fn], false, false);
  }

  // --- epilogue: C VGPR v -> rows base+v (lanes 0-15) / base+8+v (16-31) ---
  for (int fm = 0; fm < 2; ++fm)
    for (int fn = 0; fn < 4; ++fn) {
      int rbase = m0 + wm * 32 + fm * 16 + hi * 8;
      int col   = n0 + wn * 64 + fn * 16 + l;
      for (int v = 0; v < 8; ++v) {
        int row = rbase + v;
        if (row < M) C[(size_t)row * N + col] = acc[fm][fn][v];
      }
    }
}

// ---------------------------------------------------------------------------
// W[k,n] fp32  ->  Wt[n,k] bf16 (rows padded with zeros to Kp)
// ---------------------------------------------------------------------------
__global__ void convert_wt_kernel(const float* __restrict__ W,
                                  unsigned short* __restrict__ Wt,
                                  int K, int Nn, int Kp)
{
  int idx = blockIdx.x * blockDim.x + threadIdx.x;
  if (idx >= Nn * Kp) return;
  int n = idx / Kp, k = idx - n * Kp;
  Wt[idx] = (k < K) ? f32_to_bf16_bits(W[(size_t)k * Nn + n]) : (unsigned short)0;
}

// ---------------------------------------------------------------------------
// Attention score projections: es[n,h] = <h[n,h,:], a_src[h,:]>, same for ed
// ---------------------------------------------------------------------------
__global__ void escore_kernel(const float* __restrict__ h,
                              const float* __restrict__ a_src,
                              const float* __restrict__ a_dst,
                              float* __restrict__ es, float* __restrict__ ed,
                              int N, int H)
{
  int idx = blockIdx.x * blockDim.x + threadIdx.x;
  if (idx >= N * H) return;
  int n = idx / H, hd = idx - n * H;
  const float* hrow = h + (size_t)n * H * 256 + hd * 256;
  const float* as = a_src + hd * 256;
  const float* ad = a_dst + hd * 256;
  float s = 0.f, d = 0.f;
  for (int c = 0; c < 256; ++c) { float v = hrow[c]; s += v * as[c]; d += v * ad[c]; }
  es[idx] = s; ed[idx] = d;
}

__global__ void zero_attn_kernel(unsigned* mkey, float* den, int n) {
  int idx = blockIdx.x * blockDim.x + threadIdx.x;
  if (idx < n) { mkey[idx] = 0u; den[idx] = 0.f; }
}

__global__ void edge_max_kernel(const int* __restrict__ ei,
                                const float* __restrict__ es, const float* __restrict__ ed,
                                unsigned* __restrict__ mkey, int H)
{
  int idx = blockIdx.x * blockDim.x + threadIdx.x;
  if (idx >= ET_EDGES_C * H) return;
  int e = idx / H, hd = idx - e * H;
  int s = edge_src(ei, e), d = edge_dst(ei, e);
  float v = es[s * H + hd] + ed[d * H + hd];
  v = (v > 0.f) ? v : NEG_SLOPE_C * v;
  atomicMax(&mkey[d * H + hd], fkey(v));
}

__global__ void edge_sum_kernel(const int* __restrict__ ei,
                                const float* __restrict__ es, const float* __restrict__ ed,
                                const unsigned* __restrict__ mkey,
                                float* __restrict__ den, int H)
{
  int idx = blockIdx.x * blockDim.x + threadIdx.x;
  if (idx >= ET_EDGES_C * H) return;
  int e = idx / H, hd = idx - e * H;
  int s = edge_src(ei, e), d = edge_dst(ei, e);
  float v = es[s * H + hd] + ed[d * H + hd];
  v = (v > 0.f) ? v : NEG_SLOPE_C * v;
  float m = funkey(mkey[d * H + hd]);
  atomicAdd(&den[d * H + hd], __expf(v - m));
}

__global__ void fill_bias_kernel(float* __restrict__ out, const float* __restrict__ b,
                                 int nrows, int HC)
{
  long long idx = (long long)blockIdx.x * blockDim.x + threadIdx.x;
  if (idx >= (long long)nrows * HC) return;
  out[idx] = b[idx % HC];
}

// one block per edge; scatter-add weighted source features into out[dst]
__global__ void message_kernel(const int* __restrict__ ei,
                               const float* __restrict__ h,
                               const float* __restrict__ es, const float* __restrict__ ed,
                               const unsigned* __restrict__ mkey, const float* __restrict__ den,
                               float* __restrict__ out, int H)
{
  int e = blockIdx.x;
  int s = edge_src(ei, e), d = edge_dst(ei, e);
  float alpha[3];
  for (int hd = 0; hd < H; ++hd) {
    float v = es[s * H + hd] + ed[d * H + hd];
    v = (v > 0.f) ? v : NEG_SLOPE_C * v;
    float m = funkey(mkey[d * H + hd]);
    alpha[hd] = __expf(v - m) / (den[d * H + hd] + 1e-16f);
  }
  int HC = H * 256;
  for (int c = threadIdx.x; c < HC; c += blockDim.x)
    atomicAdd(&out[(size_t)d * HC + c], h[(size_t)s * HC + c] * alpha[c >> 8]);
}

// ---------------------------------------------------------------------------
// Pooling + MLP tail
// ---------------------------------------------------------------------------
__global__ void zero_pool_kernel(float* psum, float* pcnt) {
  int idx = blockIdx.x * blockDim.x + threadIdx.x;
  if (idx < N_GRAPHS_C * 256) psum[idx] = 0.f;
  if (idx < N_GRAPHS_C)       pcnt[idx] = 0.f;
}

__global__ void pool_kernel(const float* __restrict__ h, const int* __restrict__ batch,
                            float* __restrict__ psum, float* __restrict__ pcnt)
{
  int idx = blockIdx.x * blockDim.x + threadIdx.x;
  if (idx >= N_NODES_C * 256) return;
  int n = idx >> 8, c = idx & 255;
  int g = batch[n];
  atomicAdd(&psum[g * 256 + c], h[idx]);
  if (c == 0) atomicAdd(&pcnt[g], 1.0f);
}

// pooled -> z[:, 0:256]  (z stride 384)
__global__ void pool_norm_kernel(const float* __restrict__ psum, const float* __restrict__ pcnt,
                                 float* __restrict__ z)
{
  int idx = blockIdx.x * blockDim.x + threadIdx.x;
  if (idx >= N_GRAPHS_C * 256) return;
  int g = idx >> 8, c = idx & 255;
  z[g * 384 + c] = psum[idx] / fmaxf(pcnt[g], 1.0f);
}

__global__ void fc_kernel(const float* __restrict__ in, int inStride,
                          const float* __restrict__ W, const float* __restrict__ b,
                          float* __restrict__ out, int outStride,
                          int M, int N, int K, int doRelu)
{
  int idx = blockIdx.x * blockDim.x + threadIdx.x;
  if (idx >= M * N) return;
  int m = idx / N, n = idx - m * N;
  float s = b[n];
  for (int k = 0; k < K; ++k) s += in[m * inStride + k] * W[k * N + n];
  if (doRelu) s = fmaxf(s, 0.f);
  out[m * outStride + n] = s;
}

__global__ void softmax2_kernel(const float* __restrict__ logits, float* __restrict__ out) {
  int g = threadIdx.x;
  if (g >= N_GRAPHS_C) return;
  float a = logits[g * 2 + 0], b = logits[g * 2 + 1];
  float m = fmaxf(a, b);
  float ea = __expf(a - m), eb = __expf(b - m);
  float inv = 1.0f / (ea + eb);
  out[g * 2 + 0] = ea * inv;
  out[g * 2 + 1] = eb * inv;
}

// ---------------------------------------------------------------------------
// Orchestration
// ---------------------------------------------------------------------------
extern "C" void kernel_launch(void* const* d_in, const int* in_sizes, int n_in,
                              void* d_out, int out_size, void* d_ws, size_t ws_size,
                              hipStream_t stream) {
  (void)in_sizes; (void)n_in; (void)out_size; (void)ws_size;

  const float* x       = (const float*)d_in[0];
  const int*   ei      = (const int*)  d_in[1];
  const float* phy     = (const float*)d_in[2];
  const int*   batch   = (const int*)  d_in[3];
  const float* W1      = (const float*)d_in[4];
  const float* a_src1  = (const float*)d_in[5];
  const float* a_dst1  = (const float*)d_in[6];
  const float* b1      = (const float*)d_in[7];
  const float* W2      = (const float*)d_in[8];
  const float* a_src2  = (const float*)d_in[9];
  const float* a_dst2  = (const float*)d_in[10];
  const float* b2      = (const float*)d_in[11];
  const float* fc1_w1  = (const float*)d_in[12];
  const float* fc1_b1  = (const float*)d_in[13];
  const float* fc1_w2  = (const float*)d_in[14];
  const float* fc1_b2  = (const float*)d_in[15];
  const float* fc2_w1  = (const float*)d_in[16];
  const float* fc2_b1  = (const float*)d_in[17];
  const float* fc2_w2  = (const float*)d_in[18];
  const float* fc2_b2  = (const float*)d_in[19];
  float* outp = (float*)d_out;

  const int N = N_NODES_C;
  const int K1 = 1304, Kp1 = 1312, N1 = 768;
  const int K2 = 768,  Kp2 = 768,  N2 = 256;

  // workspace carve-up (256B aligned)
  char* w = (char*)d_ws;
  size_t off = 0;
  auto carve = [&](size_t bytes) -> void* {
    void* p = w + off;
    off += (bytes + 255) & ~(size_t)255;
    return p;
  };
  unsigned short* w1t   = (unsigned short*)carve((size_t)N1 * Kp1 * 2);
  unsigned short* w2t   = (unsigned short*)carve((size_t)N2 * Kp2 * 2);
  float*          h1    = (float*)carve((size_t)N * N1 * 4);   // reused as h2
  float*          out1  = (float*)carve((size_t)N * N1 * 4);   // reused as out2
  float*          es    = (float*)carve((size_t)N * 3 * 4);
  float*          ed    = (float*)carve((size_t)N * 3 * 4);
  unsigned*       mkey  = (unsigned*)carve((size_t)N * 3 * 4);
  float*          den   = (float*)carve((size_t)N * 3 * 4);
  float*          psum  = (float*)carve((size_t)N_GRAPHS_C * 256 * 4);
  float*          pcnt  = (float*)carve((size_t)N_GRAPHS_C * 4);
  float*          z     = (float*)carve((size_t)N_GRAPHS_C * 384 * 4);
  float*          t1    = (float*)carve((size_t)N_GRAPHS_C * 128 * 4);
  float*          t3    = (float*)carve((size_t)N_GRAPHS_C * 192 * 4);
  float*          logit = (float*)carve((size_t)N_GRAPHS_C * 2 * 4);
  float* h2   = h1;     // h1 dead after message1
  float* out2 = out1;   // out1 dead after gemm2 (relu-on-load)

  const int T = 256;
  auto cdiv = [](long long a, long long b) { return (unsigned)((a + b - 1) / b); };

  // weight transpose + bf16 conversion
  convert_wt_kernel<<<cdiv((long long)N1 * Kp1, T), T, 0, stream>>>(W1, w1t, K1, N1, Kp1);
  convert_wt_kernel<<<cdiv((long long)N2 * Kp2, T), T, 0, stream>>>(W2, w2t, K2, N2, Kp2);

  // ---- conv1 ----
  gemm_bf16_wmma<<<dim3(cdiv(N, BM), N1 / BN), T, 0, stream>>>(x, w1t, h1, N, N1, K1, Kp1, 0);
  escore_kernel<<<cdiv((long long)N * 3, T), T, 0, stream>>>(h1, a_src1, a_dst1, es, ed, N, 3);
  zero_attn_kernel<<<cdiv((long long)N * 3, T), T, 0, stream>>>(mkey, den, N * 3);
  edge_max_kernel<<<cdiv((long long)ET_EDGES_C * 3, T), T, 0, stream>>>(ei, es, ed, mkey, 3);
  edge_sum_kernel<<<cdiv((long long)ET_EDGES_C * 3, T), T, 0, stream>>>(ei, es, ed, mkey, den, 3);
  fill_bias_kernel<<<cdiv((long long)N * N1, T), T, 0, stream>>>(out1, b1, N, N1);
  message_kernel<<<ET_EDGES_C, T, 0, stream>>>(ei, h1, es, ed, mkey, den, out1, 3);

  // ---- conv2 (relu fused into GEMM A-load) ----
  gemm_bf16_wmma<<<dim3(cdiv(N, BM), N2 / BN), T, 0, stream>>>(out1, w2t, h2, N, N2, K2, Kp2, 1);
  escore_kernel<<<cdiv((long long)N, T), T, 0, stream>>>(h2, a_src2, a_dst2, es, ed, N, 1);
  zero_attn_kernel<<<cdiv((long long)N, T), T, 0, stream>>>(mkey, den, N);
  edge_max_kernel<<<cdiv((long long)ET_EDGES_C, T), T, 0, stream>>>(ei, es, ed, mkey, 1);
  edge_sum_kernel<<<cdiv((long long)ET_EDGES_C, T), T, 0, stream>>>(ei, es, ed, mkey, den, 1);
  fill_bias_kernel<<<cdiv((long long)N * N2, T), T, 0, stream>>>(out2, b2, N, N2);
  message_kernel<<<ET_EDGES_C, T, 0, stream>>>(ei, h2, es, ed, mkey, den, out2, 1);

  // ---- global mean pool ----
  zero_pool_kernel<<<N_GRAPHS_C, T, 0, stream>>>(psum, pcnt);
  pool_kernel<<<cdiv((long long)N * 256, T), T, 0, stream>>>(out2, batch, psum, pcnt);
  pool_norm_kernel<<<cdiv((long long)N_GRAPHS_C * 256, T), T, 0, stream>>>(psum, pcnt, z);

  // ---- phy MLP + head ----
  fc_kernel<<<cdiv(64 * 128, T), T, 0, stream>>>(phy, 188, fc1_w1, fc1_b1, t1, 128, 64, 128, 188, 1);
  fc_kernel<<<cdiv(64 * 128, T), T, 0, stream>>>(t1, 128, fc1_w2, fc1_b2, z + 256, 384, 64, 128, 128, 1);
  fc_kernel<<<cdiv(64 * 192, T), T, 0, stream>>>(z, 384, fc2_w1, fc2_b1, t3, 192, 64, 192, 384, 1);
  fc_kernel<<<cdiv(64 * 2, T), T, 0, stream>>>(t3, 192, fc2_w2, fc2_b2, logit, 2, 64, 2, 192, 0);
  softmax2_kernel<<<1, 64, 0, stream>>>(logit, outp);
}
